// Qwen3Attention_48129403519386
// MI455X (gfx1250) — matrix operand order repaired and checked by reading, not compiled, not run on hardware
//
#include <hip/hip_runtime.h>
#include <hip/hip_bf16.h>

// ---------------------------------------------------------------------------
// Qwen3 attention block for gfx1250 (MI455X): bf16 WMMA + async LDS staging.
// ---------------------------------------------------------------------------

#define B_ 2
#define T_ 2048
#define HID_ 2048
#define H_ 16
#define KV_ 8
#define D_ 128
#define NTOK 4096   // B_*T_

typedef __attribute__((ext_vector_type(16))) __bf16 v16bf;
typedef __attribute__((ext_vector_type(8)))  __bf16 v8bf;
typedef __attribute__((ext_vector_type(8)))  float  v8f;

union FragA { v16bf v; v8bf h[2]; };

#if __has_builtin(__builtin_amdgcn_global_load_async_to_lds_b128)
#define HAVE_ASYNC_LDS 1
#else
#define HAVE_ASYNC_LDS 0
#endif

// Pointer types matching the async builtin's expected parameters:
// (v4i addrspace(1)*, v4i addrspace(3)*, imm offset, imm cpol)
typedef int v4i_ __attribute__((vector_size(16)));
typedef __attribute__((address_space(1))) v4i_* as1_v4i;
typedef __attribute__((address_space(3))) v4i_* as3_v4i;
typedef __attribute__((address_space(1))) void* as1_void;
typedef __attribute__((address_space(3))) void* as3_void;

// Copy 16 bytes global -> LDS. Async (ASYNCcnt) when the gfx1250 builtin
// exists, synchronous VGPR bounce otherwise.
__device__ __forceinline__ void cp16_g2l(const __bf16* g, __bf16* l) {
#if HAVE_ASYNC_LDS
  __builtin_amdgcn_global_load_async_to_lds_b128(
      (as1_v4i)(as1_void)(void*)g,
      (as3_v4i)(as3_void)(void*)l, 0, 0);
#else
  *(uint4*)l = *(const uint4*)g;
#endif
}

__device__ __forceinline__ void async_wait0() {
#if HAVE_ASYNC_LDS
#if __has_builtin(__builtin_amdgcn_s_wait_asynccnt)
  __builtin_amdgcn_s_wait_asynccnt(0);
#else
  asm volatile("s_wait_asynccnt 0x0" ::: "memory");
#endif
#endif
}

__device__ __forceinline__ v8f zero_v8f() {
  v8f z;
  #pragma unroll
  for (int i = 0; i < 8; ++i) z[i] = 0.0f;
  return z;
}

__device__ __forceinline__ v8f wmma_bf16(v16bf a, v16bf b, v8f c) {
  return __builtin_amdgcn_wmma_f32_16x16x32_bf16(false, a, false, b, (short)0, c,
                                                 false, false);
}

// ---------------------------------------------------------------------------
// f32 -> bf16 elementwise convert
// ---------------------------------------------------------------------------
__global__ void cvt_f32_bf16(const float* __restrict__ s, __bf16* __restrict__ d,
                             long n) {
  long i = (long)blockIdx.x * blockDim.x + threadIdx.x;
  if (i < n) d[i] = (__bf16)s[i];
}

// W (K x N) f32 -> Wt (N x K) bf16   (so GEMM B-tile staging is a straight copy)
__global__ void cvt_w_transpose(const float* __restrict__ s, __bf16* __restrict__ d,
                                int K, int N) {
  long i = (long)blockIdx.x * 256 + threadIdx.x;   // row-major over K x N
  int n = (int)(i % N);
  int k = (int)(i / N);
  d[(long)n * K + k] = (__bf16)s[i];
}

// v: [B,T,KV,D] f32 -> [B,KV,D,T] bf16 (d-major, so attention V staging is a copy)
__global__ void vcvt_transpose(const float* __restrict__ s, __bf16* __restrict__ d) {
  long i = (long)blockIdx.x * 256 + threadIdx.x;   // n = B_*T_*KV_*D_
  int  dd = (int)(i & 127);
  long r  = i >> 7;          // (b*T + t)*KV + kv
  int  kv = (int)(r & 7);
  long bt = r >> 3;
  int  t  = (int)(bt & (T_ - 1));
  int  b  = (int)(bt >> 11);
  d[(((long)(b * KV_ + kv) * D_ + dd) * T_) + t] = (__bf16)s[i];
}

// ---------------------------------------------------------------------------
// Tiled bf16 WMMA GEMM:  C(MxN,f32) = A(MxK,bf16,row) * Bt(NxK,bf16,row)
// Block tile 128x128, BK=32, double-buffered async LDS staging,
// 8 waves of 32x64 (2x4 WMMA tiles each), one barrier per K-step.
// ---------------------------------------------------------------------------
__global__ __launch_bounds__(256) void gemm_bf16_wmma(
    const __bf16* __restrict__ A, const __bf16* __restrict__ Bt,
    float* __restrict__ C, int M, int N, int K) {
  __shared__ __attribute__((aligned(16))) __bf16 As[2][128][32];
  __shared__ __attribute__((aligned(16))) __bf16 Bs[2][128][32];

  const int tid = threadIdx.x, wid = tid >> 5, lane = tid & 31;
  const int lane16 = lane & 15, g = lane >> 4;
  const int nblk = N / 128;
  const int m0 = (blockIdx.x / nblk) * 128;
  const int n0 = (blockIdx.x % nblk) * 128;
  const int wm = wid >> 1;   // 0..3 : 32-row band
  const int wn = wid & 1;    // 0..1 : 64-col band

  // staging assignment: each thread copies one 16-elem chunk of A and of Bt
  const int srow = tid >> 1, shf = tid & 1;
  const __bf16* aSrc = A  + (long)(m0 + srow) * K + shf * 16;
  const __bf16* bSrc = Bt + (long)(n0 + srow) * K + shf * 16;

  v8f acc[2][4];
  #pragma unroll
  for (int s = 0; s < 2; ++s)
    #pragma unroll
    for (int t = 0; t < 4; ++t) acc[s][t] = zero_v8f();

  auto stage = [&](int k0, int buf) {
    cp16_g2l(aSrc + k0,     &As[buf][srow][shf * 16]);
    cp16_g2l(aSrc + k0 + 8, &As[buf][srow][shf * 16 + 8]);
    cp16_g2l(bSrc + k0,     &Bs[buf][srow][shf * 16]);
    cp16_g2l(bSrc + k0 + 8, &Bs[buf][srow][shf * 16 + 8]);
  };

  stage(0, 0);
  const int NK = K >> 5;
  for (int kt = 0; kt < NK; ++kt) {
    const int cur = kt & 1;
    async_wait0();
    __syncthreads();                         // buf[cur] ready; buf[cur^1] drained
    if (kt + 1 < NK) stage((kt + 1) << 5, cur ^ 1);

    FragA af[2];
    #pragma unroll
    for (int s = 0; s < 2; ++s) {
      int r = wm * 32 + s * 16 + lane16;
      af[s].h[0] = *(const v8bf*)&As[cur][r][8 * g];        // K = 8g..8g+7
      af[s].h[1] = *(const v8bf*)&As[cur][r][16 + 8 * g];   // K = 16+8g..
    }
    #pragma unroll
    for (int t = 0; t < 4; ++t) {
      FragA bf;
      int col = wn * 64 + t * 16 + lane16;
      bf.h[0] = *(const v8bf*)&Bs[cur][col][8 * g];         // K = 8g..8g+7
      bf.h[1] = *(const v8bf*)&Bs[cur][col][16 + 8 * g];
      #pragma unroll
      for (int s = 0; s < 2; ++s)
        acc[s][t] = wmma_bf16(af[s].v, bf.v, acc[s][t]);
    }
  }

  #pragma unroll
  for (int s = 0; s < 2; ++s)
    #pragma unroll
    for (int t = 0; t < 4; ++t)
      #pragma unroll
      for (int r = 0; r < 8; ++r) {
        int row = m0 + wm * 32 + s * 16 + r + 8 * g;   // C layout: M = r + 8*half
        int col = n0 + wn * 64 + t * 16 + lane16;
        C[(long)row * N + col] = acc[s][t][r];
      }
}

// ---------------------------------------------------------------------------
// Fused RMSNorm + RoPE.  src f32 [B,T,NH,D] -> dst bf16 [B,NH,T,D]
// One wave per (b,t,head) row; 4 d-values per lane.
// ---------------------------------------------------------------------------
__global__ __launch_bounds__(256) void rmsnorm_rope(
    const float* __restrict__ src, const float* __restrict__ w,
    const float* __restrict__ cosb, const float* __restrict__ sinb,
    __bf16* __restrict__ dst, int NH) {
  const int lane = threadIdx.x & 31, wid = threadIdx.x >> 5;
  long row = (long)blockIdx.x * 8 + wid;     // over B*T*NH
  int  hh  = (int)(row % NH);
  long bt  = row / NH;                        // b*T + t
  int  t   = (int)(bt & (T_ - 1));
  int  b   = (int)(bt >> 11);

  const float* x = src + row * D_;
  float xi[4], xn[4];
  float ss = 0.f;
  #pragma unroll
  for (int i = 0; i < 4; ++i) {
    int dd = 4 * lane + i;
    xi[i] = x[dd];
    ss += xi[i] * xi[i];
  }
  #pragma unroll
  for (int off = 16; off >= 1; off >>= 1) ss += __shfl_xor(ss, off, 32);
  float rinv = rsqrtf(ss * (1.f / 128.f) + 1e-6f);

  const float* cp = cosb + (bt << 7);
  const float* sp = sinb + (bt << 7);
  __bf16* op = dst + (((long)b * NH + hh) * T_ + t) * D_;

  #pragma unroll
  for (int i = 0; i < 4; ++i) {
    int dd = 4 * lane + i;
    xn[i] = xi[i] * rinv * w[dd];
  }
  #pragma unroll
  for (int i = 0; i < 4; ++i) {
    int dd = 4 * lane + i;
    float other = __shfl_xor(xn[i], 16, 32);            // x[(d+64)%128]
    float rot = (lane < 16) ? -other : other;           // rotate_half
    op[dd] = (__bf16)(xn[i] * cp[dd] + rot * sp[dd]);
  }
}

// ---------------------------------------------------------------------------
// Causal GQA flash attention, bf16 WMMA, fp32 online softmax.
// One block = 128 queries of one (b,h); each wave owns 16 queries.
// Double-buffered async staging of 32-key K/V tiles.
// ---------------------------------------------------------------------------
__global__ __launch_bounds__(256) void attn_wmma(
    const __bf16* __restrict__ qb, const __bf16* __restrict__ kb,
    const __bf16* __restrict__ vb, __bf16* __restrict__ ob) {
  __shared__ __attribute__((aligned(16))) __bf16 Ks[2][32][128];   // [key][d]
  __shared__ __attribute__((aligned(16))) __bf16 Vt[2][128][32];   // [d][key]
  __shared__ float Pb[8][16][32];                                  // per-wave P bounce

  const int tid = threadIdx.x, wid = tid >> 5, lane = tid & 31;
  const int lane16 = lane & 15, g = lane >> 4;

  const int blk  = blockIdx.x;                 // B*H*(T/128) blocks
  const int qblk = blk % (T_ / 128);
  const int h    = (blk / (T_ / 128)) % H_;
  const int b    = blk / ((T_ / 128) * H_);
  const int hkv  = h >> 1;                     // H/KV = 2
  const int q0   = qblk * 128 + wid * 16;
  const int qend = q0 + 15;

  const __bf16* qbase = qb + (((long)b * H_ + h) * T_ + q0) * D_;
  const __bf16* kbase = kb + ((long)b * KV_ + hkv) * T_ * D_;   // [t][d]
  const __bf16* vbase = vb + ((long)b * KV_ + hkv) * D_ * T_;   // [d][t]

  // Q fragments (A layout), 4 chunks of K=32 over D=128
  FragA qf[4];
  #pragma unroll
  for (int kc = 0; kc < 4; ++kc) {
    const __bf16* p = qbase + (long)lane16 * D_ + kc * 32;
    qf[kc].h[0] = *(const v8bf*)(p + 16 * g);
    qf[kc].h[1] = *(const v8bf*)(p + 16 * g + 8);
  }

  v8f o[8];
  float m8[8], l8[8];
  #pragma unroll
  for (int c = 0; c < 8; ++c) o[c] = zero_v8f();
  #pragma unroll
  for (int r = 0; r < 8; ++r) { m8[r] = -1e30f; l8[r] = 0.f; }

  const float scale = 0.08838834764831845f;    // 1/sqrt(128)
  const int kmax = qblk * 128 + 127;           // block-wide causal bound

  // staging assignment
  const int skk = tid >> 3, sdg = tid & 7;     // K tile: key row, d-group
  const int svd = tid >> 1, svh = tid & 1;     // V tile: d row, key half
  auto stage = [&](int k0, int buf) {
    const __bf16* ks = kbase + (long)(k0 + skk) * D_ + sdg * 16;
    cp16_g2l(ks,     &Ks[buf][skk][sdg * 16]);
    cp16_g2l(ks + 8, &Ks[buf][skk][sdg * 16 + 8]);
    const __bf16* vs = vbase + (long)svd * T_ + k0 + svh * 16;
    cp16_g2l(vs,     &Vt[buf][svd][svh * 16]);
    cp16_g2l(vs + 8, &Vt[buf][svd][svh * 16 + 8]);
  };

  stage(0, 0);
  int it = 0;
  for (int k0 = 0; k0 <= kmax; k0 += 32, ++it) {
    const int cur = it & 1;
    async_wait0();
    __syncthreads();
    if (k0 + 32 <= kmax) stage(k0 + 32, cur ^ 1);
    if (k0 > qend) continue;   // wave-uniform causal skip

    // ---- S = Q * K^T  (two 16x16 key halves) ----
    v8f s0 = zero_v8f(), s1 = zero_v8f();
    #pragma unroll
    for (int kc = 0; kc < 4; ++kc) {
      FragA kf0, kf1;
      kf0.h[0] = *(const v8bf*)&Ks[cur][lane16][kc * 32 + 16 * g];
      kf0.h[1] = *(const v8bf*)&Ks[cur][lane16][kc * 32 + 16 * g + 8];
      kf1.h[0] = *(const v8bf*)&Ks[cur][16 + lane16][kc * 32 + 16 * g];
      kf1.h[1] = *(const v8bf*)&Ks[cur][16 + lane16][kc * 32 + 16 * g + 8];
      s0 = wmma_bf16(qf[kc].v, kf0.v, s0);
      s1 = wmma_bf16(qf[kc].v, kf1.v, s1);
    }

    // ---- online softmax (C layout rows: M = r + 8g, N = lane16) ----
    #pragma unroll
    for (int r = 0; r < 8; ++r) {
      int qg = q0 + r + 8 * g;
      float a0 = s0[r] * scale + ((k0 + lane16) > qg ? -1e9f : 0.f);
      float a1 = s1[r] * scale + ((k0 + 16 + lane16) > qg ? -1e9f : 0.f);
      float mx = fmaxf(a0, a1);
      #pragma unroll
      for (int off = 8; off >= 1; off >>= 1) mx = fmaxf(mx, __shfl_xor(mx, off, 32));
      float mnew = fmaxf(m8[r], mx);
      float sc = __expf(m8[r] - mnew);
      float p0 = __expf(a0 - mnew);
      float p1 = __expf(a1 - mnew);
      float ps = p0 + p1;
      #pragma unroll
      for (int off = 8; off >= 1; off >>= 1) ps += __shfl_xor(ps, off, 32);
      l8[r] = l8[r] * sc + ps;
      m8[r] = mnew;
      #pragma unroll
      for (int c = 0; c < 8; ++c) o[c][r] = o[c][r] * sc;
      Pb[wid][r + 8 * g][lane16]      = p0;
      Pb[wid][r + 8 * g][16 + lane16] = p1;
    }
    asm volatile("s_wait_dscnt 0x0" ::: "memory");   // wave-local LDS bounce fence

    // ---- O += P * V ----
    FragA pf;
    #pragma unroll
    for (int j = 0; j < 8; ++j) pf.h[0][j] = (__bf16)Pb[wid][lane16][8 * g + j];
    #pragma unroll
    for (int j = 0; j < 8; ++j) pf.h[1][j] = (__bf16)Pb[wid][lane16][16 + 8 * g + j];
    #pragma unroll
    for (int c = 0; c < 8; ++c) {
      FragA vf;
      vf.h[0] = *(const v8bf*)&Vt[cur][c * 16 + lane16][8 * g];
      vf.h[1] = *(const v8bf*)&Vt[cur][c * 16 + lane16][16 + 8 * g];
      o[c] = wmma_bf16(pf.v, vf.v, o[c]);
    }
  }

  // epilogue: normalize and write [tok, h*128+d] bf16
  #pragma unroll
  for (int c = 0; c < 8; ++c) {
    #pragma unroll
    for (int r = 0; r < 8; ++r) {
      int qrow = q0 + r + 8 * g;
      float val = o[c][r] / l8[r];
      ob[((long)b * T_ + qrow) * (H_ * D_) + h * D_ + c * 16 + lane16] = (__bf16)val;
    }
  }
}

// ---------------------------------------------------------------------------
// Host side
// ---------------------------------------------------------------------------
extern "C" void kernel_launch(void* const* d_in, const int* in_sizes, int n_in,
                              void* d_out, int out_size, void* d_ws, size_t ws_size,
                              hipStream_t stream) {
  (void)in_sizes; (void)n_in; (void)out_size; (void)ws_size;
  const float* hs   = (const float*)d_in[0];
  const float* cosb = (const float*)d_in[1];
  const float* sinb = (const float*)d_in[2];
  /* d_in[3] = attention_mask: causal triu(-1e9) computed analytically */
  const float* Wq = (const float*)d_in[4];
  const float* Wk = (const float*)d_in[5];
  const float* Wv = (const float*)d_in[6];
  const float* Wo = (const float*)d_in[7];
  const float* qw = (const float*)d_in[8];
  const float* kw = (const float*)d_in[9];
  float* out = (float*)d_out;

  char* ws = (char*)d_ws;
  size_t off = 0;
  auto carve = [&](size_t bytes) -> void* {
    void* p = ws + off;
    off += (bytes + 255) & ~(size_t)255;
    return p;
  };
  __bf16* Xbf   = (__bf16*)carve((size_t)NTOK * HID_ * 2);
  __bf16* Wqt   = (__bf16*)carve((size_t)HID_ * (H_ * D_) * 2);   // [N][K]
  __bf16* Wkt   = (__bf16*)carve((size_t)HID_ * (KV_ * D_) * 2);
  __bf16* Wvt   = (__bf16*)carve((size_t)HID_ * (KV_ * D_) * 2);
  __bf16* Wot   = (__bf16*)carve((size_t)(H_ * D_) * HID_ * 2);
  float*  qraw  = (float*)carve((size_t)NTOK * (H_ * D_) * 4);
  float*  kraw  = (float*)carve((size_t)NTOK * (KV_ * D_) * 4);
  float*  vraw  = (float*)carve((size_t)NTOK * (KV_ * D_) * 4);
  __bf16* qbf   = (__bf16*)carve((size_t)B_ * H_ * T_ * D_ * 2);
  __bf16* kbf   = (__bf16*)carve((size_t)B_ * KV_ * T_ * D_ * 2);
  __bf16* vbf   = (__bf16*)carve((size_t)B_ * KV_ * T_ * D_ * 2);
  __bf16* attnb = (__bf16*)carve((size_t)NTOK * (H_ * D_) * 2);

  // 1) convert activations + (transposed) weights to bf16
  {
    long n = (long)NTOK * HID_;
    cvt_f32_bf16<<<dim3((n + 255) / 256), dim3(256), 0, stream>>>(hs, Xbf, n);
  }
  cvt_w_transpose<<<dim3((HID_ * H_ * D_) / 256), dim3(256), 0, stream>>>(
      Wq, Wqt, HID_, H_ * D_);
  cvt_w_transpose<<<dim3((HID_ * KV_ * D_) / 256), dim3(256), 0, stream>>>(
      Wk, Wkt, HID_, KV_ * D_);
  cvt_w_transpose<<<dim3((HID_ * KV_ * D_) / 256), dim3(256), 0, stream>>>(
      Wv, Wvt, HID_, KV_ * D_);
  cvt_w_transpose<<<dim3(((H_ * D_) * HID_) / 256), dim3(256), 0, stream>>>(
      Wo, Wot, H_ * D_, HID_);

  // 2) QKV projections (WMMA GEMMs)
  gemm_bf16_wmma<<<dim3((NTOK / 128) * ((H_ * D_) / 128)), dim3(256), 0, stream>>>(
      Xbf, Wqt, qraw, NTOK, H_ * D_, HID_);
  gemm_bf16_wmma<<<dim3((NTOK / 128) * ((KV_ * D_) / 128)), dim3(256), 0, stream>>>(
      Xbf, Wkt, kraw, NTOK, KV_ * D_, HID_);
  gemm_bf16_wmma<<<dim3((NTOK / 128) * ((KV_ * D_) / 128)), dim3(256), 0, stream>>>(
      Xbf, Wvt, vraw, NTOK, KV_ * D_, HID_);

  // 3) RMSNorm + RoPE (q, k); V layout convert (d-major)
  rmsnorm_rope<<<dim3((B_ * T_ * H_) / 8), dim3(256), 0, stream>>>(
      qraw, qw, cosb, sinb, qbf, H_);
  rmsnorm_rope<<<dim3((B_ * T_ * KV_) / 8), dim3(256), 0, stream>>>(
      kraw, kw, cosb, sinb, kbf, KV_);
  vcvt_transpose<<<dim3((B_ * T_ * KV_ * D_) / 256), dim3(256), 0, stream>>>(vraw, vbf);

  // 4) causal GQA flash attention
  attn_wmma<<<dim3(B_ * H_ * (T_ / 128)), dim3(256), 0, stream>>>(qbf, kbf, vbf, attnb);

  // 5) output projection -> fp32 d_out
  gemm_bf16_wmma<<<dim3((NTOK / 128) * (HID_ / 128)), dim3(256), 0, stream>>>(
      attnb, Wot, out, NTOK, HID_, H_ * D_);
}
